// DualQuantizer_85392539779411
// MI455X (gfx1250) — matrix operand order repaired and checked by reading, not compiled, not run on hardware
//
#include <hip/hip_runtime.h>
#include <hip/hip_bf16.h>
#include <cstdint>

// ---------------- problem constants ----------------
#define B_    8
#define T_    2048
#define N_    (B_ * T_)      // 16384 rows of z_flat
#define D_    256            // SEM_DIM
#define K_    8192           // codebook entries
#define DTOT  292
#define AC_   36
#define HALF_LVL 10.0f       // (N_LEVELS-1)/2

// ---------------- d_out layout (float elements, return order) ----------------
#define OUT_Z     0
#define OUT_IDX   (B_ * DTOT * T_)            // 4,784,128
#define OUT_AC    (OUT_IDX + N_)              // 4,800,512
#define OUT_LOSS  (OUT_AC + B_ * AC_ * T_)    // 5,390,336
#define OUT_ZACQ  (OUT_LOSS + 1)              // 5,390,337

// ---------------- d_ws layout (bytes) ----------------
#define WS_AH   ((size_t)0)
#define WS_AL   (WS_AH + (size_t)N_ * D_ * 2)     // +8 MB
#define WS_BH   (WS_AL + (size_t)N_ * D_ * 2)     // +8 MB
#define WS_BL   (WS_BH + (size_t)K_ * D_ * 2)     // +4 MB
#define WS_CN   (WS_BL + (size_t)K_ * D_ * 2)     // +4 MB
#define WS_LOSS (WS_CN + (size_t)K_ * 4)          // +32 KB

typedef __attribute__((ext_vector_type(16))) _Float16 v16h;
typedef __attribute__((ext_vector_type(8)))  _Float16 v8h;
typedef __attribute__((ext_vector_type(8)))  float    v8f;

// ============================================================
// init: zero the loss accumulator (re-run every launch)
// ============================================================
__global__ void k_init(float* loss) {
    if (threadIdx.x == 0) *loss = 0.0f;
}

// ============================================================
// pack A: z[:, :256, :] -> row-major (N,256) f16 hi/lo split
// n = b*T + t ; element (n,d) reads z[b, d, t]
// ============================================================
__global__ __launch_bounds__(256) void k_pack_a(const float* __restrict__ z,
                                                _Float16* __restrict__ ah,
                                                _Float16* __restrict__ al) {
    int gid = blockIdx.x * 256 + threadIdx.x;   // N_*D_ threads; gid = n*256+d
    int n = gid >> 8, d = gid & 255;
    int b = n >> 11, t = n & 2047;
    float x = z[((size_t)b * DTOT + d) * T_ + t];
    _Float16 h = (_Float16)x;
    _Float16 l = (_Float16)(x - (float)h);
    ah[gid] = h;
    al[gid] = l;
}

// ============================================================
// pack B: codebook (K,256) f32 -> f16 hi/lo split (row-major kept)
// ============================================================
__global__ __launch_bounds__(256) void k_pack_b(const float* __restrict__ cb,
                                                _Float16* __restrict__ bh,
                                                _Float16* __restrict__ bl) {
    int gid = blockIdx.x * 256 + threadIdx.x;   // K_*D_ threads
    float x = cb[gid];
    _Float16 h = (_Float16)x;
    _Float16 l = (_Float16)(x - (float)h);
    bh[gid] = h;
    bl[gid] = l;
}

// ============================================================
// cnorm[k] = ||codebook[k]||^2 in exact f32
// ============================================================
__global__ __launch_bounds__(256) void k_cnorm(const float* __restrict__ cb,
                                               float* __restrict__ cn) {
    int k = blockIdx.x * 256 + threadIdx.x;     // K_ threads
    const float4* p = (const float4*)(cb + (size_t)k * D_);
    float s = 0.0f;
    #pragma unroll 8
    for (int i = 0; i < D_ / 4; ++i) {
        float4 v = p[i];
        s += v.x * v.x + v.y * v.y + v.z * v.z + v.w * v.w;
    }
    cn[k] = s;
}

// ============================================================
// Fused GEMM (f16 hi/lo split, V_WMMA_F32_16X16X32_F16) + argmin.
// Block = 256 threads = 8 waves; wave w owns rows n0..n0+15.
// A frags (16x256 hi+lo) live in registers for the whole K loop.
// For each 16-code chunk: 8 k-steps x 3 WMMAs (AhBh, AhBl, AlBh)
// on 3 independent accumulator chains, then fold cn - 2*dot into
// a running (min,idx). Final 16-lane shfl_xor argmin reduce.
// out_idx must point at the indices region of d_out.
// ============================================================
__global__ __launch_bounds__(256) void k_gemm_argmin(
        const _Float16* __restrict__ Ah, const _Float16* __restrict__ Al,
        const _Float16* __restrict__ Bh, const _Float16* __restrict__ Bl,
        const float*    __restrict__ cn, float* __restrict__ out_idx) {
    const int wave = threadIdx.x >> 5;
    const int lane = threadIdx.x & 31;
    const int m    = lane & 15;     // A row within tile / C col / B col
    const int half = lane >> 4;     // lane half selects K sub-range
    const int n0   = (blockIdx.x * 8 + wave) * 16;

    union U { v16h v; v8h h[2]; };

    // ---- load A fragments once (contiguous k order, same as B:
    //      lane<16: K = {s*32+0..15}; lane>=16: +16) ----
    v16h ah[8], al[8];
    {
        const _Float16* arh = Ah + (size_t)(n0 + m) * D_;
        const _Float16* arl = Al + (size_t)(n0 + m) * D_;
        #pragma unroll
        for (int s = 0; s < 8; ++s) {
            int base = s * 32 + half * 16;
            U u; u.h[0] = *(const v8h*)(arh + base);
                 u.h[1] = *(const v8h*)(arh + base + 8); ah[s] = u.v;
            U w; w.h[0] = *(const v8h*)(arl + base);
                 w.h[1] = *(const v8h*)(arl + base + 8); al[s] = w.v;
        }
    }

    float minv[8];
    int   mini[8];
    #pragma unroll
    for (int r = 0; r < 8; ++r) { minv[r] = 3.4e38f; mini[r] = 0; }

    for (int kb = 0; kb < K_; kb += 16) {
        // B in the same contiguous k order as A: lane holds N = lane&15, K = (lane>>4)*16 + 0..15
        // -> 32 contiguous bytes of the codebook row per k-step.
        const _Float16* bhp = Bh + (size_t)(kb + m) * D_ + half * 16;
        const _Float16* blp = Bl + (size_t)(kb + m) * D_ + half * 16;
        v8f a0 = {}, a1 = {}, a2 = {};
        #pragma unroll
        for (int s = 0; s < 8; ++s) {
            v16h bh = *(const v16h*)(bhp + s * 32);
            v16h bl = *(const v16h*)(blp + s * 32);
            a0 = __builtin_amdgcn_wmma_f32_16x16x32_f16(false, ah[s], false, bh,
                                                        (short)0, a0, false, false);
            a1 = __builtin_amdgcn_wmma_f32_16x16x32_f16(false, ah[s], false, bl,
                                                        (short)0, a1, false, false);
            a2 = __builtin_amdgcn_wmma_f32_16x16x32_f16(false, al[s], false, bh,
                                                        (short)0, a2, false, false);
        }
        float c = cn[kb + m];
        #pragma unroll
        for (int r = 0; r < 8; ++r) {
            float dot = a0[r] + a1[r] + a2[r];
            float sd  = fmaf(-2.0f, dot, c);        // dist minus row-constant ||z||^2
            if (sd < minv[r]) { minv[r] = sd; mini[r] = kb + m; }
        }
    }

    // ---- argmin reduce across the 16 lanes of each half (C cols),
    //      tie-break to the smallest index (jnp.argmin semantics) ----
    #pragma unroll
    for (int r = 0; r < 8; ++r) {
        #pragma unroll
        for (int off = 1; off < 16; off <<= 1) {
            float ov = __shfl_xor(minv[r], off, 32);
            int   oi = __shfl_xor(mini[r], off, 32);
            if (ov < minv[r] || (ov == minv[r] && oi < mini[r])) {
                minv[r] = ov; mini[r] = oi;
            }
        }
    }
    // C layout: VGPR r, lanes 0-15 -> row r; lanes 16-31 -> row r+8
    if (m == 0) {
        #pragma unroll
        for (int r = 0; r < 8; ++r)
            out_idx[n0 + r + 8 * half] = (float)mini[r];
    }
}

// ============================================================
// Epilogue: z_out[:, :256, :] = codebook[idx] (exact f32 gather),
// accumulate sum((z_q - z_sem)^2) for vq_loss.
// idxf must point at the indices region of d_out.
// ============================================================
__global__ __launch_bounds__(256) void k_epilogue(const float* __restrict__ z,
                                                  const float* __restrict__ cb,
                                                  const float* __restrict__ idxf,
                                                  float* __restrict__ out,
                                                  float* __restrict__ loss) {
    int n = blockIdx.x;          // row
    int d = threadIdx.x;         // channel
    int b = n >> 11, t = n & 2047;
    int idx = (int)idxf[n];
    idx = idx < 0 ? 0 : (idx > K_ - 1 ? K_ - 1 : idx);   // defensive clamp
    float c  = cb[(size_t)idx * D_ + d];
    size_t zo = ((size_t)b * DTOT + d) * T_ + t;
    float zs = z[zo];
    out[zo] = c;                 // z_sem_q == z_q numerically
    float df = c - zs;

    __shared__ float red[256];
    red[d] = df * df;
    __syncthreads();
    #pragma unroll
    for (int s = 128; s > 0; s >>= 1) {
        if (d < s) red[d] += red[d + s];
        __syncthreads();
    }
    if (d == 0) atomicAdd(loss, red[0]);
}

// ============================================================
// AC path: z_b = tanh(x)*10; z_ac_q = round(z_b);
// ac_codes = clip(round(z_ac_q + 10), 0, 20)
// ============================================================
__global__ __launch_bounds__(256) void k_ac(const float* __restrict__ z,
                                            float* __restrict__ out) {
    int gid = blockIdx.x * 256 + threadIdx.x;   // B_*AC_*T_ threads
    int t = gid & 2047;
    int rest = gid >> 11;
    int d = rest % AC_;
    int b = rest / AC_;
    float x = z[((size_t)b * DTOT + D_ + d) * T_ + t];
    float q = rintf(tanhf(x) * HALF_LVL);                 // round-half-even
    float code = fminf(fmaxf(rintf(q + HALF_LVL), 0.0f), 20.0f);
    out[((size_t)b * DTOT + D_ + d) * T_ + t] = q;        // z_out AC block
    out[OUT_AC + gid]   = code;                           // ac_codes (as f32)
    out[OUT_ZACQ + gid] = q;                              // z_ac_q
}

// ============================================================
// vq_loss = (1 + BETA) * mean((z_q - z_sem)^2)
// ============================================================
__global__ void k_finalize(const float* __restrict__ loss, float* __restrict__ out) {
    out[OUT_LOSS] = 1.1f * (*loss) / (float)((size_t)N_ * D_);
}

// ============================================================
extern "C" void kernel_launch(void* const* d_in, const int* in_sizes, int n_in,
                              void* d_out, int out_size, void* d_ws, size_t ws_size,
                              hipStream_t stream) {
    const float* z  = (const float*)d_in[0];   // (8, 292, 2048) f32
    const float* cb = (const float*)d_in[1];   // (8192, 256) f32
    float* out = (float*)d_out;

    char* ws = (char*)d_ws;
    _Float16* Ah = (_Float16*)(ws + WS_AH);
    _Float16* Al = (_Float16*)(ws + WS_AL);
    _Float16* Bh = (_Float16*)(ws + WS_BH);
    _Float16* Bl = (_Float16*)(ws + WS_BL);
    float*    Cn = (float*)   (ws + WS_CN);
    float*    Ls = (float*)   (ws + WS_LOSS);

    k_init<<<1, 32, 0, stream>>>(Ls);
    k_pack_a<<<(N_ * D_) / 256, 256, 0, stream>>>(z, Ah, Al);
    k_pack_b<<<(K_ * D_) / 256, 256, 0, stream>>>(cb, Bh, Bl);
    k_cnorm<<<K_ / 256, 256, 0, stream>>>(cb, Cn);
    k_gemm_argmin<<<N_ / 128, 256, 0, stream>>>(Ah, Al, Bh, Bl, Cn, out + OUT_IDX);
    k_epilogue<<<N_, 256, 0, stream>>>(z, cb, out + OUT_IDX, out, Ls);
    k_ac<<<(B_ * AC_ * T_) / 256, 256, 0, stream>>>(z, out);
    k_finalize<<<1, 1, 0, stream>>>(Ls, out);
}
